// AttentionWithFeedForward_36558761623966
// MI455X (gfx1250) — matrix operand order repaired and checked by reading, not compiled, not run on hardware
//
#include <hip/hip_runtime.h>

// ---------------------------------------------------------------------------
// Types / helpers
// ---------------------------------------------------------------------------
using bf16 = __bf16;
typedef __attribute__((ext_vector_type(16))) __bf16 v16bf;
typedef __attribute__((ext_vector_type(8)))  __bf16 v8bf;
typedef __attribute__((ext_vector_type(8)))  float  v8f;

__device__ __forceinline__ bf16 f2bf(float f) {
    unsigned int u = __float_as_uint(f);
    u += 0x7FFFu + ((u >> 16) & 1u);           // round-to-nearest-even
    unsigned short s = (unsigned short)(u >> 16);
    return __builtin_bit_cast(bf16, s);
}

// Generic pointers into LDS carry the LDS byte offset in their low 32 bits
// (ISA 10.2 aperture mapping); async VDST operand wants that 32-bit address.
__device__ __forceinline__ unsigned lds_off_u32(const void* p) {
    return (unsigned)(unsigned long long)p;
}

// CDNA5 async global->LDS copy (ASYNCcnt path), CDNA5 ISA
__device__ __forceinline__ void async_load_b128(const void* gptr, unsigned ldsoff) {
    asm volatile("global_load_async_to_lds_b128 %0, %1, off"
                 :: "v"(ldsoff), "v"((unsigned long long)gptr)
                 : "memory");
}
__device__ __forceinline__ void wait_async0() {
    asm volatile("s_wait_asynccnt 0x0" ::: "memory");
}
// async loads complete in order per wave: <=3 outstanding => oldest tile landed
__device__ __forceinline__ void wait_async3() {
    asm volatile("s_wait_asynccnt 0x3" ::: "memory");
}

__device__ __forceinline__ v16bf frag16(const bf16* p0, const bf16* p1) {
    v8bf lo = *(const v8bf*)p0;
    v8bf hi = *(const v8bf*)p1;
    v16bf r;
    #pragma unroll
    for (int i = 0; i < 8; ++i) { r[i] = lo[i]; r[i + 8] = hi[i]; }
    return r;
}

#define LDSK 48   // 32 + 16 pad: row stride 96B keeps ds_load_b128 alignment

// ---------------------------------------------------------------------------
// fp32 -> bf16 elementwise conversion
// ---------------------------------------------------------------------------
__global__ void cvt_f32_bf16(const float* __restrict__ src, bf16* __restrict__ dst, long n) {
    long i = (long)blockIdx.x * blockDim.x + threadIdx.x;
    long stride = (long)gridDim.x * blockDim.x;
    for (; i < n; i += stride) dst[i] = f2bf(src[i]);
}

// fp32 [R][C] -> bf16 [C][R] transpose (R, C multiples of 32)
__global__ __launch_bounds__(256)
void cvt_transpose(const float* __restrict__ S, bf16* __restrict__ D, int R, int C) {
    __shared__ float t[32][33];
    const int tx = threadIdx.x, ty = threadIdx.y;       // block (32,8)
    const long r0 = (long)blockIdx.x * 32, c0 = (long)blockIdx.y * 32;
    #pragma unroll
    for (int i = 0; i < 4; ++i)
        t[ty + i * 8][tx] = S[(r0 + ty + i * 8) * C + c0 + tx];
    __syncthreads();
    #pragma unroll
    for (int i = 0; i < 4; ++i)
        D[(c0 + ty + i * 8) * R + r0 + tx] = f2bf(t[tx][ty + i * 8]);
}

// bf16 [R][C] (ld_s, batched) -> bf16 [C][Rpad] (zero-padded rows R..Rpad)
__global__ __launch_bounds__(256)
void transpose_pad_bf16(const bf16* __restrict__ S, bf16* __restrict__ D,
                        int R, int C, long ld_s, int Rpad,
                        int zdiv, long sSb, long sSh, long sD) {
    __shared__ bf16 t[32][33];
    const int z = blockIdx.z, bb = z / zdiv, hh = z % zdiv;
    S += bb * sSb + hh * sSh;
    D += (long)z * sD;
    const int tx = threadIdx.x, ty = threadIdx.y;
    const long r0 = (long)blockIdx.x * 32, c0 = (long)blockIdx.y * 32;
    #pragma unroll
    for (int i = 0; i < 4; ++i) {
        long r = r0 + ty + i * 8;
        t[ty + i * 8][tx] = (r < R) ? S[r * ld_s + c0 + tx] : f2bf(0.0f);
    }
    __syncthreads();
    #pragma unroll
    for (int i = 0; i < 4; ++i) {
        long r = r0 + tx;
        if (r < Rpad) D[(c0 + ty + i * 8) * (long)Rpad + r] = t[tx][ty + i * 8];
    }
}

// ---------------------------------------------------------------------------
// ALIGNED WMMA GEMM (M%128==0, N%64==0, K%32==0), B stored [N][K].
// Block tile 128x64, 8 wave32 (4x2), wave tile 32x32 -> 4 WMMA / K-step.
// Double-buffered LDS fed by GLOBAL_LOAD_ASYNC_TO_LDS_B128: tile i+1's HBM
// latency hides under tile i's WMMAs (in-order async completion per wave).
// ---------------------------------------------------------------------------
template<int RELU, int OUT_BF16, int OUT_F32>
__global__ __launch_bounds__(256)
void gemm_async(const bf16* __restrict__ A, const bf16* __restrict__ B,
                const float* __restrict__ bias,
                float* __restrict__ Cf, bf16* __restrict__ Cb,
                int M, int N, int K, long lda, long ldb, long ldc,
                int zdiv, long sAb, long sAh, long sBb, long sBh, long sCb, long sCh)
{
    __shared__ bf16 As[2][128][LDSK];
    __shared__ bf16 Bs[2][64][LDSK];
    (void)M; (void)N;

    const int z = blockIdx.z, bb = z / zdiv, hh = z % zdiv;
    A += bb * sAb + hh * sAh;
    B += bb * sBb + hh * sBh;
    const long coff = bb * sCb + hh * sCh;

    const int tid = threadIdx.x, lane = tid & 31, wave = tid >> 5;
    const int wm = wave >> 1, wn = wave & 1;
    const long m0 = (long)blockIdx.x * 128, n0 = (long)blockIdx.y * 64;

    const int sr = tid >> 2;              // staging row 0..63
    const int sk = (tid & 3) * 8;         // staging k-chunk (8 bf16 = 16B)
    const bf16* pA0 = &A[(m0 + sr) * lda + sk];
    const bf16* pA1 = pA0 + 64 * lda;
    const bf16* pB  = &B[(n0 + sr) * ldb + sk];

    v8f acc[4];
    #pragma unroll
    for (int i = 0; i < 4; ++i)
        #pragma unroll
        for (int j = 0; j < 8; ++j) acc[i][j] = 0.0f;

    // prologue: tile 0 -> buffer 0
    async_load_b128(pA0, lds_off_u32(&As[0][sr][sk]));
    async_load_b128(pA1, lds_off_u32(&As[0][sr + 64][sk]));
    async_load_b128(pB,  lds_off_u32(&Bs[0][sr][sk]));

    int buf = 0;
    for (int kt = 0; kt < K; kt += 32) {
        if (kt + 32 < K) {   // issue tile i+1 into the other buffer
            const int nb = buf ^ 1;
            async_load_b128(pA0 + kt + 32, lds_off_u32(&As[nb][sr][sk]));
            async_load_b128(pA1 + kt + 32, lds_off_u32(&As[nb][sr + 64][sk]));
            async_load_b128(pB  + kt + 32, lds_off_u32(&Bs[nb][sr][sk]));
            wait_async3();   // tile i (3 oldest) has landed
        } else {
            wait_async0();
        }
        __syncthreads();     // all waves see tile i

        const int fr = lane & 15, fh = lane >> 4;
        v16bf a0 = frag16(&As[buf][wm * 32 + fr][fh * 8],
                          &As[buf][wm * 32 + fr][16 + fh * 8]);
        v16bf a1 = frag16(&As[buf][wm * 32 + 16 + fr][fh * 8],
                          &As[buf][wm * 32 + 16 + fr][16 + fh * 8]);
        v16bf b0 = frag16(&Bs[buf][wn * 32 + fr][fh * 8],
                          &Bs[buf][wn * 32 + fr][16 + fh * 8]);
        v16bf b1 = frag16(&Bs[buf][wn * 32 + 16 + fr][fh * 8],
                          &Bs[buf][wn * 32 + 16 + fr][16 + fh * 8]);
        acc[0] = __builtin_amdgcn_wmma_f32_16x16x32_bf16(false, a0, false, b0, (short)0, acc[0], false, false);
        acc[1] = __builtin_amdgcn_wmma_f32_16x16x32_bf16(false, a0, false, b1, (short)0, acc[1], false, false);
        acc[2] = __builtin_amdgcn_wmma_f32_16x16x32_bf16(false, a1, false, b0, (short)0, acc[2], false, false);
        acc[3] = __builtin_amdgcn_wmma_f32_16x16x32_bf16(false, a1, false, b1, (short)0, acc[3], false, false);
        __syncthreads();     // all waves done reading buf before it is re-filled
        buf ^= 1;
    }

    const int fr = lane & 15, fh = lane >> 4;
    #pragma unroll
    for (int mi = 0; mi < 2; ++mi)
        #pragma unroll
        for (int ni = 0; ni < 2; ++ni) {
            v8f a = acc[mi * 2 + ni];
            const long gc = n0 + wn * 32 + ni * 16 + fr;
            const float bv = bias ? bias[gc] : 0.0f;
            #pragma unroll
            for (int v = 0; v < 8; ++v) {
                const long gr = m0 + wm * 32 + mi * 16 + fh * 8 + v;
                float val = a[v] + bv;
                if (RELU) val = fmaxf(val, 0.0f);
                if (OUT_F32) Cf[coff + gr * ldc + gc] = val;
                if (OUT_BF16) Cb[coff + gr * ldc + gc] = f2bf(val);
            }
        }
}

// ---------------------------------------------------------------------------
// SAFE WMMA GEMM (guarded tails), B stored [N][K]. Block tile 64x64.
// Used only for the small ragged GEMMs (M=308 / N=77).
// ---------------------------------------------------------------------------
template<int RELU, int OUT_BF16, int OUT_F32>
__global__ __launch_bounds__(256)
void gemm_safe(const bf16* __restrict__ A, const bf16* __restrict__ B,
               const float* __restrict__ bias,
               float* __restrict__ Cf, bf16* __restrict__ Cb,
               int M, int N, int K, long lda, long ldb, long ldc,
               int zdiv, long sAb, long sAh, long sBb, long sBh, long sCb, long sCh)
{
    __shared__ bf16 As[64][LDSK];
    __shared__ bf16 Bs[64][LDSK];

    const int z = blockIdx.z, bb = z / zdiv, hh = z % zdiv;
    A += bb * sAb + hh * sAh;
    B += bb * sBb + hh * sBh;
    const long coff = bb * sCb + hh * sCh;

    const int tid = threadIdx.x, lane = tid & 31, wave = tid >> 5;
    const int wm = wave >> 1, wn = wave & 1;
    const int m0 = blockIdx.x * 64, n0 = blockIdx.y * 64;

    v8f acc0, acc1;
    #pragma unroll
    for (int j = 0; j < 8; ++j) { acc0[j] = 0.0f; acc1[j] = 0.0f; }

    const int sr = tid >> 2;
    const int sk = (tid & 3) * 8;

    for (int kt = 0; kt < K; kt += 32) {
        {   // A tile
            const int gr = m0 + sr;
            bf16* dst = &As[sr][sk];
            if (gr < M && (kt + sk + 7) < K) {
                *(v8bf*)dst = *(const v8bf*)&A[(long)gr * lda + kt + sk];
            } else {
                #pragma unroll
                for (int i = 0; i < 8; ++i) {
                    int kk = kt + sk + i;
                    dst[i] = (gr < M && kk < K) ? A[(long)gr * lda + kk] : f2bf(0.0f);
                }
            }
        }
        {   // B tile ([N][K] straight copy)
            const int gn = n0 + sr;
            bf16* dst = &Bs[sr][sk];
            if (gn < N && (kt + sk + 7) < K) {
                *(v8bf*)dst = *(const v8bf*)&B[(long)gn * ldb + kt + sk];
            } else {
                #pragma unroll
                for (int i = 0; i < 8; ++i) {
                    int kk = kt + sk + i;
                    dst[i] = (gn < N && kk < K) ? B[(long)gn * ldb + kk] : f2bf(0.0f);
                }
            }
        }
        __syncthreads();

        const int fr = lane & 15, fh = lane >> 4;
        v16bf a  = frag16(&As[wm * 16 + fr][fh * 8],       &As[wm * 16 + fr][16 + fh * 8]);
        v16bf b0 = frag16(&Bs[wn * 32 + fr][fh * 8],      &Bs[wn * 32 + fr][16 + fh * 8]);
        v16bf b1 = frag16(&Bs[wn * 32 + 16 + fr][fh * 8], &Bs[wn * 32 + 16 + fr][16 + fh * 8]);
        acc0 = __builtin_amdgcn_wmma_f32_16x16x32_bf16(false, a, false, b0, (short)0, acc0, false, false);
        acc1 = __builtin_amdgcn_wmma_f32_16x16x32_bf16(false, a, false, b1, (short)0, acc1, false, false);
        __syncthreads();
    }

    const int fr = lane & 15, fh = lane >> 4;
    #pragma unroll
    for (int j = 0; j < 2; ++j) {
        v8f a = j ? acc1 : acc0;
        const int gc = n0 + wn * 32 + j * 16 + fr;
        if (gc >= N) continue;
        const float bv = bias ? bias[gc] : 0.0f;
        #pragma unroll
        for (int v = 0; v < 8; ++v) {
            const int gr = m0 + wm * 16 + fh * 8 + v;
            if (gr >= M) continue;
            float val = a[v] + bv;
            if (RELU) val = fmaxf(val, 0.0f);
            if (OUT_F32) Cf[coff + (long)gr * ldc + gc] = val;
            if (OUT_BF16) Cb[coff + (long)gr * ldc + gc] = f2bf(val);
        }
    }
}

// ---------------------------------------------------------------------------
// Row softmax (scale folded in): f32 scores -> bf16 probs, zero-pads to ld_out
// ---------------------------------------------------------------------------
__global__ __launch_bounds__(256)
void softmax_rows(const float* __restrict__ S, bf16* __restrict__ P,
                  int cols, long ld_in, long ld_out, float scale)
{
    const long row = blockIdx.x;
    const float* s = S + row * ld_in;
    bf16* p = P + row * ld_out;
    __shared__ float red[256];
    const int tid = threadIdx.x;

    float m = -3.0e38f;
    for (int c = tid; c < cols; c += 256) m = fmaxf(m, s[c] * scale);
    red[tid] = m; __syncthreads();
    for (int off = 128; off > 0; off >>= 1) {
        if (tid < off) red[tid] = fmaxf(red[tid], red[tid + off]);
        __syncthreads();
    }
    m = red[0]; __syncthreads();

    float sum = 0.0f;
    for (int c = tid; c < cols; c += 256) sum += __expf(s[c] * scale - m);
    red[tid] = sum; __syncthreads();
    for (int off = 128; off > 0; off >>= 1) {
        if (tid < off) red[tid] += red[tid + off];
        __syncthreads();
    }
    const float inv = 1.0f / red[0];

    for (int c = tid; c < cols; c += 256) p[c] = f2bf(__expf(s[c] * scale - m) * inv);
    for (int c = cols + tid; c < (int)ld_out; c += 256) p[c] = f2bf(0.0f);
}

// ---------------------------------------------------------------------------
// residual + LayerNorm (1024 cols)
// ---------------------------------------------------------------------------
__global__ __launch_bounds__(256)
void resid_ln(const float* __restrict__ X, const float* __restrict__ Dl,
              const float* __restrict__ g, const float* __restrict__ b,
              float* __restrict__ Of, bf16* __restrict__ Ob)
{
    const long row = blockIdx.x;
    const float* x = X  + row * 1024;
    const float* d = Dl + row * 1024;
    __shared__ float r1[256];
    __shared__ float r2[256];
    const int tid = threadIdx.x;

    float hv[4];
    float s = 0.0f, s2 = 0.0f;
    #pragma unroll
    for (int i = 0; i < 4; ++i) {
        const int c = tid + i * 256;
        const float h = x[c] + d[c];
        hv[i] = h; s += h; s2 += h * h;
    }
    r1[tid] = s; r2[tid] = s2; __syncthreads();
    for (int off = 128; off > 0; off >>= 1) {
        if (tid < off) { r1[tid] += r1[tid + off]; r2[tid] += r2[tid + off]; }
        __syncthreads();
    }
    const float mu  = r1[0] * (1.0f / 1024.0f);
    const float var = r2[0] * (1.0f / 1024.0f) - mu * mu;
    const float rs  = rsqrtf(var + 1e-5f);
    #pragma unroll
    for (int i = 0; i < 4; ++i) {
        const int c = tid + i * 256;
        const float o = (hv[i] - mu) * rs * g[c] + b[c];
        Of[row * 1024 + c] = o;
        if (Ob) Ob[row * 1024 + c] = f2bf(o);
    }
}

// ---------------------------------------------------------------------------
// Launch
// ---------------------------------------------------------------------------
extern "C" void kernel_launch(void* const* d_in, const int* in_sizes, int n_in,
                              void* d_out, int out_size, void* d_ws, size_t ws_size,
                              hipStream_t stream) {
    (void)in_sizes; (void)n_in; (void)out_size; (void)ws_size;

    const float* x     = (const float*)d_in[0];
    const float* y     = (const float*)d_in[1];
    const float* w_qkv = (const float*)d_in[2];
    const float* b_qkv = (const float*)d_in[3];
    const float* w_so  = (const float*)d_in[4];
    const float* b_so  = (const float*)d_in[5];
    const float* w_q   = (const float*)d_in[6];
    const float* b_q   = (const float*)d_in[7];
    const float* w_k   = (const float*)d_in[8];
    const float* b_k   = (const float*)d_in[9];
    const float* w_v   = (const float*)d_in[10];
    const float* b_v   = (const float*)d_in[11];
    const float* w_co  = (const float*)d_in[12];
    const float* b_co  = (const float*)d_in[13];
    const float* w1    = (const float*)d_in[14];
    const float* b1    = (const float*)d_in[15];
    const float* w2    = (const float*)d_in[16];
    const float* b2    = (const float*)d_in[17];
    const float* w3    = (const float*)d_in[18];
    const float* b3    = (const float*)d_in[19];
    const float* ln_g  = (const float*)d_in[20];
    const float* ln_b  = (const float*)d_in[21];

    char* base = (char*)d_ws;
    size_t cur = 0;
    auto alloc = [&](size_t bytes) -> void* {
        void* p = base + cur;
        cur = (cur + bytes + 255) & ~(size_t)255;
        return p;
    };
    const long M4 = 4096;
    bf16* Xbf   = (bf16*)alloc(M4 * 1024 * 2);
    bf16* Ybf   = (bf16*)alloc(4 * 77 * 768 * 2);
    bf16* WqkvT = (bf16*)alloc(3072L * 1024 * 2);   // [N][K]
    bf16* WsoT  = (bf16*)alloc(1024L * 1024 * 2);
    bf16* WqT   = (bf16*)alloc(1024L * 1024 * 2);
    bf16* WkT   = (bf16*)alloc(1024L * 768 * 2);
    bf16* WvT   = (bf16*)alloc(1024L * 768 * 2);
    bf16* WcoT  = (bf16*)alloc(1024L * 1024 * 2);
    bf16* W1T   = (bf16*)alloc(4096L * 1024 * 2);
    bf16* W2T   = (bf16*)alloc(4096L * 4096 * 2);
    bf16* W3T   = (bf16*)alloc(1024L * 4096 * 2);
    bf16* QKV   = (bf16*)alloc(M4 * 3072 * 2);
    bf16* VT    = (bf16*)alloc(64L * 64 * 1024 * 2);   // self V^T per (b,h)
    float* Sc   = (float*)alloc(64L * 1024 * 1024 * 4);
    bf16* Ps    = (bf16*)alloc(64L * 1024 * 1024 * 2);
    bf16* AO    = (bf16*)alloc(M4 * 1024 * 2);
    float* SA   = (float*)alloc(M4 * 1024 * 4);
    float* X1   = (float*)alloc(M4 * 1024 * 4);
    bf16* X1b   = (bf16*)alloc(M4 * 1024 * 2);
    bf16* QC    = (bf16*)alloc(M4 * 1024 * 2);
    bf16* KC    = (bf16*)alloc(308L * 1024 * 2);
    bf16* VC    = (bf16*)alloc(308L * 1024 * 2);
    bf16* VTc   = (bf16*)alloc(64L * 64 * 96 * 2);     // cross V^T, K padded to 96
    float* ScC  = (float*)alloc(64L * 1024 * 96 * 4);
    bf16* PC    = (bf16*)alloc(64L * 1024 * 96 * 2);
    bf16* AOC   = (bf16*)alloc(M4 * 1024 * 2);
    float* CA   = (float*)alloc(M4 * 1024 * 4);
    float* X2   = (float*)alloc(M4 * 1024 * 4);
    bf16* X2b   = (bf16*)alloc(M4 * 1024 * 2);
    bf16* H1    = (bf16*)alloc(M4 * 4096 * 2);
    bf16* H2    = (bf16*)alloc(M4 * 4096 * 2);
    float* FFo  = (float*)alloc(M4 * 1024 * 4);

    auto cvt = [&](const float* s, bf16* d, long n) {
        long g = (n + 255) / 256; if (g > 16384) g = 16384;
        cvt_f32_bf16<<<dim3((unsigned)g), 256, 0, stream>>>(s, d, n);
    };
    cvt(x, Xbf, M4 * 1024);
    cvt(y, Ybf, 4 * 77 * 768);
    const dim3 tb(32, 8);
    cvt_transpose<<<dim3(32, 96),  tb, 0, stream>>>(w_qkv, WqkvT, 1024, 3072);
    cvt_transpose<<<dim3(32, 32),  tb, 0, stream>>>(w_so,  WsoT,  1024, 1024);
    cvt_transpose<<<dim3(32, 32),  tb, 0, stream>>>(w_q,   WqT,   1024, 1024);
    cvt_transpose<<<dim3(24, 32),  tb, 0, stream>>>(w_k,   WkT,    768, 1024);
    cvt_transpose<<<dim3(24, 32),  tb, 0, stream>>>(w_v,   WvT,    768, 1024);
    cvt_transpose<<<dim3(32, 32),  tb, 0, stream>>>(w_co,  WcoT,  1024, 1024);
    cvt_transpose<<<dim3(32, 128), tb, 0, stream>>>(w1,    W1T,   1024, 4096);
    cvt_transpose<<<dim3(128,128), tb, 0, stream>>>(w2,    W2T,   4096, 4096);
    cvt_transpose<<<dim3(128, 32), tb, 0, stream>>>(w3,    W3T,   4096, 1024);

    const float iss = 0.125f;   // 1/sqrt(64)

    // 1) QKV = x @ w_qkv + b_qkv  -> bf16
    gemm_async<0,1,0><<<dim3(32,48,1),256,0,stream>>>(Xbf, WqkvT, b_qkv, nullptr, QKV,
        4096,3072,1024, 1024,1024,3072, 1, 0,0,0,0,0,0);
    // V^T slices for attn@V
    transpose_pad_bf16<<<dim3(32,2,64), tb, 0, stream>>>(QKV + 2048, VT,
        1024, 64, 3072, 1024, 16, 1024L*3072, 64, 64L*1024);
    // 2) self scores = q @ k^T (k rows ARE [N][K])
    gemm_async<0,0,1><<<dim3(8,16,64),256,0,stream>>>(QKV, QKV + 1024, nullptr, Sc, nullptr,
        1024,1024,64, 3072,3072,1024, 16,
        1024L*3072, 64, 1024L*3072, 64, 16L*1048576, 1048576L);
    // 3) softmax
    softmax_rows<<<dim3(65536),256,0,stream>>>(Sc, Ps, 1024, 1024, 1024, iss);
    // 4) attn @ v
    gemm_async<0,1,0><<<dim3(8,1,64),256,0,stream>>>(Ps, VT, nullptr, nullptr, AO,
        1024,64,1024, 1024,1024,1024, 16,
        16L*1048576, 1048576L, 16L*65536, 65536L, 1048576L, 64);
    // 5) self out-proj -> f32
    gemm_async<0,0,1><<<dim3(32,16,1),256,0,stream>>>(AO, WsoT, b_so, SA, nullptr,
        4096,1024,1024, 1024,1024,1024, 1, 0,0,0,0,0,0);
    // 6) x1 = LN(x + sa)
    resid_ln<<<dim3(4096),256,0,stream>>>(x, SA, ln_g, ln_b, X1, X1b);
    // 7) cross q
    gemm_async<0,1,0><<<dim3(32,16,1),256,0,stream>>>(X1b, WqT, b_q, nullptr, QC,
        4096,1024,1024, 1024,1024,1024, 1, 0,0,0,0,0,0);
    // 8-9) cross k, v (M=308 -> guarded)
    gemm_safe<0,1,0><<<dim3(5,16,1),256,0,stream>>>(Ybf, WkT, b_k, nullptr, KC,
        308,1024,768, 768,768,1024, 1, 0,0,0,0,0,0);
    gemm_safe<0,1,0><<<dim3(5,16,1),256,0,stream>>>(Ybf, WvT, b_v, nullptr, VC,
        308,1024,768, 768,768,1024, 1, 0,0,0,0,0,0);
    // cross V^T slices, K padded 77 -> 96 with zeros
    transpose_pad_bf16<<<dim3(3,2,64), tb, 0, stream>>>(VC, VTc,
        77, 64, 1024, 96, 16, 77L*1024, 64, 64L*96);
    // 10) cross scores (N=77 -> guarded)
    gemm_safe<0,0,1><<<dim3(16,2,64),256,0,stream>>>(QC, KC, nullptr, ScC, nullptr,
        1024,77,64, 1024,1024,96, 16,
        1048576L, 64, 77L*1024, 64, 16L*98304, 98304L);
    // 11) softmax, zero-pads probs cols 77..95
    softmax_rows<<<dim3(65536),256,0,stream>>>(ScC, PC, 77, 96, 96, iss);
    // 12) cross attn @ v  (K=96, padding contributes zero)
    gemm_async<0,1,0><<<dim3(8,1,64),256,0,stream>>>(PC, VTc, nullptr, nullptr, AOC,
        1024,64,96, 96,96,1024, 16,
        16L*98304, 98304L, 16L*6144, 6144L, 1048576L, 64);
    // 13) cross out-proj -> f32
    gemm_async<0,0,1><<<dim3(32,16,1),256,0,stream>>>(AOC, WcoT, b_co, CA, nullptr,
        4096,1024,1024, 1024,1024,1024, 1, 0,0,0,0,0,0);
    // 14) x2 = LN(x1 + ca)
    resid_ln<<<dim3(4096),256,0,stream>>>(X1, CA, ln_g, ln_b, X2, X2b);
    // 15-17) FFN
    gemm_async<1,1,0><<<dim3(32,64,1),256,0,stream>>>(X2b, W1T, b1, nullptr, H1,
        4096,4096,1024, 1024,1024,4096, 1, 0,0,0,0,0,0);
    gemm_async<1,1,0><<<dim3(32,64,1),256,0,stream>>>(H1, W2T, b2, nullptr, H2,
        4096,4096,4096, 4096,4096,4096, 1, 0,0,0,0,0,0);
    gemm_async<0,0,1><<<dim3(32,16,1),256,0,stream>>>(H2, W3T, b3, FFo, nullptr,
        4096,1024,4096, 4096,4096,1024, 1, 0,0,0,0,0,0);
    // 18) out = LN(x2 + ff)
    resid_ln<<<dim3(4096),256,0,stream>>>(X2, FFo, ln_g, ln_b, (float*)d_out, nullptr);
}
